// PatchedSegmentationMapPredictor_9680856285724
// MI455X (gfx1250) — compile-verified, matrix operand order checked by reading, not run on hardware
//
#include <hip/hip_runtime.h>
#include <hip/hip_bf16.h>
#include <hip/hip_fp16.h>

#define D_MODEL   128
#define N_HEADS   8
#define HEAD_DIM  16
#define FFN_DIM   512
#define K_TOT     164
#define K_PAD     176
#define M_TILES   11
#define LDS_ROW   136   // padded halfs per feats row (272B => 4-bank stagger per row)

#define USE_ASYNC_LOGITS 1

typedef _Float16 v16h __attribute__((ext_vector_type(16)));
typedef _Float16 v8h  __attribute__((ext_vector_type(8)));
typedef _Float16 v4h  __attribute__((ext_vector_type(4)));
typedef float    v8f  __attribute__((ext_vector_type(8)));

#define INV_2PI 0.15915494309f
#define SCORE_UPSCALE 64.f     // keeps packed f16 partials in normal range

// 10^(-j/4), j=0..3  (THETA=10, nf=HEAD_DIM/4=4)
__device__ __forceinline__ float inv_freq(int j) {
    const float tbl[4] = {1.0f, 0.56234133f, 0.31622777f, 0.17782794f};
    return tbl[j & 3];
}

__device__ __forceinline__ float fast_tanh(float z) {
    const float az = fabsf(z);
    const float e  = __expf(2.f * az);          // v_exp_f32
    const float t  = 1.f - 2.f / (e + 1.f);
    return copysignf(t, z);
}

// LayerNorm stats over v128[0..127], block-wide (256 threads, waves 4..7 contribute 0)
__device__ __forceinline__ void block_ln_stats(const float* v128, float* red, int t,
                                               float& mean, float& rstd) {
    int lane = t & 31, wid = t >> 5;
    float v  = (t < 128) ? v128[t] : 0.f;
    float v2 = v * v;
    #pragma unroll
    for (int off = 16; off; off >>= 1) {
        v  += __shfl_down(v,  off, 32);
        v2 += __shfl_down(v2, off, 32);
    }
    if (lane == 0) { red[wid] = v; red[8 + wid] = v2; }
    __syncthreads();
    if (t == 0) {
        float s = 0.f, s2 = 0.f;
        for (int w = 0; w < 8; ++w) { s += red[w]; s2 += red[8 + w]; }
        float m = s * (1.f / 128.f);
        red[16] = m;
        red[17] = rsqrtf(s2 * (1.f / 128.f) - m * m + 1e-5f);
    }
    __syncthreads();
    mean = red[16];
    rstd = red[17];
}

__device__ __forceinline__ v16h ld_afrag(const _Float16* rp) {
    v8h lo = *(const v8h*)rp;          // ds_load_b128
    v8h hh = *(const v8h*)(rp + 16);   // ds_load_b128
    return __builtin_shufflevector(lo, hh, 0, 1, 2, 3, 4, 5, 6, 7,
                                   8, 9, 10, 11, 12, 13, 14, 15);
}

// Relative-rope score epilogue.  D holds P_i (even cols) / C_i (odd cols) with q,
// 1/sqrt(d) and SCORE_UPSCALE folded into B'.  score = sum_i cosD*P_i + sinD*C_i.
// Branchless trig: odd lanes get cos(x - pi/2) == sin(x) via -0.25 turn offset
// folded into qangn; v_cos_f32 consumes turns (1/2pi pre-folded into fln/qangn).
// Two rows are packed as f16x2 per butterfly to halve the ds_bpermute chain.
__device__ __forceinline__ void k_epilogue(const v8f& acc, int mt, int hi, int nn,
                                           float fln, float qangn, int ax,
                                           const float (*kpos_sh)[2],
                                           const float* vmask_sh, float* scores_row) {
    #pragma unroll
    for (int r = 0; r < 8; r += 2) {
        const int m = mt * 16 + r + hi * 8;               // rows m, m+1
        const float u0 = fmaf(kpos_sh[m][ax],     fln, qangn);
        const float u1 = fmaf(kpos_sh[m + 1][ax], fln, qangn);
        const float t0 = __builtin_amdgcn_cosf(u0);       // v_cos_f32 (turns)
        const float t1 = __builtin_amdgcn_cosf(u1);
        __half2 pv = __floats2half2_rn(acc[r] * t0, acc[r + 1] * t1);
        #pragma unroll
        for (int off = 1; off <= 8; off <<= 1) {
            int iv = __builtin_bit_cast(int, pv);
            iv = __shfl_xor(iv, off, 32);
            pv = __hadd2(pv, __builtin_bit_cast(__half2, iv));
        }
        if (nn == 0) {
            float s0 = __low2float(pv)  * (1.f / SCORE_UPSCALE);
            float s1 = __high2float(pv) * (1.f / SCORE_UPSCALE);
            if (vmask_sh[m] == 0.f)     s0 = -1e9f;
            if (vmask_sh[m + 1] == 0.f) s1 = -1e9f;
            scores_row[m]     = s0;
            scores_row[m + 1] = s1;
        }
    }
}

// One transformer layer for one query per workgroup (256 threads = 8 waves, wave w = head w).
__global__ __launch_bounds__(256)
void layer_kernel(const float* __restrict__ qpos,     // (N,2)
                  const int*   __restrict__ boffs,    // (nb+1)
                  const float* __restrict__ fm,       // (B,256,256,4,128)
                  const int*   __restrict__ lshapes,  // (4,2)
                  const float* __restrict__ Wq, const float* __restrict__ Wk,
                  const float* __restrict__ Wv, const float* __restrict__ Wo,
                  const float* __restrict__ g1, const float* __restrict__ b1v,
                  const float* __restrict__ g2, const float* __restrict__ b2v,
                  const float* __restrict__ W1, const float* __restrict__ b1f,
                  const float* __restrict__ W2, const float* __restrict__ b2f,
                  float* __restrict__ x_ws, int nb) {
    __shared__ __align__(16) _Float16 feats_sh[K_PAD][LDS_ROW];
    __shared__ float kpos_sh[K_PAD][2];
    __shared__ float vmask_sh[K_PAD];
    __shared__ int   roff_sh[K_PAD];
    __shared__ float scores_sh[N_HEADS][K_PAD];   // scores; later reused as split-K partials
    __shared__ float x_sh[128], h_sh[128], q_sh[128], o_sh[128];
    __shared__ float g_sh[1024];        // attn@feats (8x128); reused as FFN activations (512)
    __shared__ float red_sh[20];

    const int n    = blockIdx.x;
    const int t    = threadIdx.x;
    const int lane = t & 31;
    const int wid  = t >> 5;

    const float px = qpos[2 * n + 0];
    const float py = qpos[2 * n + 1];
    int bid = 0;
    for (int b = 1; b < nb; ++b) if (n >= boffs[b]) bid = b;

    __builtin_prefetch(Wk, 0, 3);
    __builtin_prefetch(W1, 0, 3);
    __builtin_prefetch(W2, 0, 3);

    // ---- per-key setup: level decode, clipped gather address, validity, key pos ----
    if (t < K_PAD) {
        if (t < K_TOT) {
            const int l    = (t < 9) ? 0 : (t < 34) ? 1 : (t < 83) ? 2 : 3;
            const int kst  = (l == 0) ? 0 : (l == 1) ? 9 : (l == 2) ? 34 : 83;
            const int d    = 3 + 2 * l;
            const int idx  = t - kst;
            const int di   = idx / d - d / 2;
            const int dj   = idx % d - d / 2;
            const int Hl   = lshapes[2 * l + 0];
            const int Wl   = lshapes[2 * l + 1];
            const float mH = (float)lshapes[0], mW = (float)lshapes[1];
            const float sH = (float)Hl / mH, sW = (float)Wl / mW;
            const int ci = (int)floorf(px * sH), cj = (int)floorf(py * sW);
            const int gi = ci + di, gj = cj + dj;
            const bool ok = (gi >= 0) && (gi < Hl) && (gj >= 0) && (gj < Wl);
            int cci = gi < 0 ? 0 : (gi > Hl - 1 ? Hl - 1 : gi);
            int ccj = gj < 0 ? 0 : (gj > Wl - 1 ? Wl - 1 : gj);
            roff_sh[t]     = (((bid * 256 + cci) * 256 + ccj) * 4 + l) * 128;
            vmask_sh[t]    = ok ? 1.f : 0.f;
            kpos_sh[t][0]  = ((float)gi + 0.5f) / sH;
            kpos_sh[t][1]  = ((float)gj + 0.5f) / sW;
        } else {
            vmask_sh[t] = 0.f; roff_sh[t] = 0;
            kpos_sh[t][0] = 0.f; kpos_sh[t][1] = 0.f;
        }
    }
    if (t < 128) x_sh[t] = x_ws[n * 128 + t];
    __syncthreads();

    // ---- gather patch features -> LDS as f16 (zeroed when invalid) ----
    for (int c = t; c < K_TOT * 32; c += 256) {
        const int kk = c >> 5, part = c & 31;
        const float m = vmask_sh[kk];
        const float4 f = *(const float4*)(fm + roff_sh[kk] + part * 4);
        v4h hv = {(_Float16)(f.x * m), (_Float16)(f.y * m),
                  (_Float16)(f.z * m), (_Float16)(f.w * m)};
        *(v4h*)&feats_sh[kk][part * 4] = hv;
    }
    for (int c = t; c < (K_PAD - K_TOT) * 32; c += 256) {   // zero pad rows 164..175
        const int kk = K_TOT + (c >> 5), part = c & 31;
        v4h z = {};
        *(v4h*)&feats_sh[kk][part * 4] = z;
    }

    // ---- LN1 ----
    float mean, rstd;
    block_ln_stats(x_sh, red_sh, t, mean, rstd);
    if (t < 128) h_sh[t] = (x_sh[t] - mean) * rstd * g1[t] + b1v[t];
    __syncthreads();

    // ---- q = h @ Wq (no rope needed: relative rope is folded into B') ----
    if (t < 128) {
        float a = 0.f;
        for (int c = 0; c < 128; ++c) a += h_sh[c] * Wq[c * 128 + t];
        q_sh[t] = a;
    }
    __syncthreads();

    // ---- WMMA B' fragments: Wk columns combined with q (scale folded), 4 k-steps ----
    const int hi = lane >> 4;         // half-wave select
    const int nn = lane & 15;         // A row within tile / B column within tile
    v16h bfrag[4];
    {
        const int pr = nn >> 1;                    // rope pair within head
        const int c0 = wid * 16 + 2 * pr;          // source Wk columns c0, c0+1
        const float p0 = q_sh[c0], p1 = q_sh[c0 + 1];
        const float sc = 0.25f * SCORE_UPSCALE;    // 1/sqrt(d) * f16-range upscale
        const float cA = sc * ((nn & 1) ? p1 : p0);
        const float cB = sc * ((nn & 1) ? -p0 : p1);
        #pragma unroll
        for (int s = 0; s < 4; ++s) {
            const int kb = s * 32 + hi * 8;
            v16h bf;
            #pragma unroll
            for (int j = 0; j < 8; ++j) {
                const int k0 = kb + j, k1 = kb + 16 + j;
                bf[j]     = (_Float16)(Wk[k0 * 128 + c0] * cA + Wk[k0 * 128 + c0 + 1] * cB);
                bf[j + 8] = (_Float16)(Wk[k1 * 128 + c0] * cA + Wk[k1 * 128 + c0 + 1] * cB);
            }
            bfrag[s] = bf;
        }
    }

    // per-lane rope constants (turns; odd lanes shifted -0.25 turn => sin)
    const int   pi_   = nn >> 1;
    const float fr    = inv_freq(pi_ & 3);
    const float fln   = fr * INV_2PI;
    const int   ax    = pi_ >> 2;
    const float qangn = -((pi_ & 4) ? py : px) * fr * INV_2PI
                        - ((nn & 1) ? 0.25f : 0.f);

    // ---- K projection GEMM (176x128 @ 128x16 per head), 2 M-tiles in flight ----
    for (int mtp = 0; mtp < 5; ++mtp) {
        const int mt0 = 2 * mtp, mt1 = mt0 + 1;
        v8f acc0 = {0.f, 0.f, 0.f, 0.f, 0.f, 0.f, 0.f, 0.f};
        v8f acc1 = {0.f, 0.f, 0.f, 0.f, 0.f, 0.f, 0.f, 0.f};
        #pragma unroll
        for (int s = 0; s < 4; ++s) {
            const int kb = s * 32 + hi * 8;
            v16h af0 = ld_afrag(&feats_sh[mt0 * 16 + nn][kb]);
            v16h af1 = ld_afrag(&feats_sh[mt1 * 16 + nn][kb]);
            acc0 = __builtin_amdgcn_wmma_f32_16x16x32_f16(
                false, af0, false, bfrag[s], (short)0, acc0, false, false);
            acc1 = __builtin_amdgcn_wmma_f32_16x16x32_f16(
                false, af1, false, bfrag[s], (short)0, acc1, false, false);
        }
        k_epilogue(acc0, mt0, hi, nn, fln, qangn, ax, kpos_sh, vmask_sh, scores_sh[wid]);
        k_epilogue(acc1, mt1, hi, nn, fln, qangn, ax, kpos_sh, vmask_sh, scores_sh[wid]);
    }
    {   // tail tile mt=10
        v8f acc = {0.f, 0.f, 0.f, 0.f, 0.f, 0.f, 0.f, 0.f};
        #pragma unroll
        for (int s = 0; s < 4; ++s) {
            const int kb = s * 32 + hi * 8;
            v16h af = ld_afrag(&feats_sh[10 * 16 + nn][kb]);
            acc = __builtin_amdgcn_wmma_f32_16x16x32_f16(
                false, af, false, bfrag[s], (short)0, acc, false, false);
        }
        k_epilogue(acc, 10, hi, nn, fln, qangn, ax, kpos_sh, vmask_sh, scores_sh[wid]);
    }
    __syncthreads();

    // ---- softmax over keys (wave w = head w) ----
    {
        float mx = -1e30f;
        for (int k = lane; k < K_TOT; k += 32) mx = fmaxf(mx, scores_sh[wid][k]);
        #pragma unroll
        for (int off = 16; off; off >>= 1) mx = fmaxf(mx, __shfl_xor(mx, off, 32));
        float sum = 0.f;
        for (int k = lane; k < K_TOT; k += 32) {
            const float e = __expf(scores_sh[wid][k] - mx);
            scores_sh[wid][k] = e;
            sum += e;
        }
        #pragma unroll
        for (int off = 16; off; off >>= 1) sum += __shfl_xor(sum, off, 32);
        const float inv = 1.f / sum;
        for (int k = lane; k < K_TOT; k += 32) scores_sh[wid][k] *= inv;
    }
    __syncthreads();

    // ---- g[h] = attn[h] @ feats (vectorized: 4 consecutive f16 cols per lane) ----
    {
        const int c4 = lane * 4;
        float a0 = 0.f, a1 = 0.f, a2 = 0.f, a3 = 0.f;
        for (int k = 0; k < K_TOT; ++k) {
            const float s = scores_sh[wid][k];
            v4h f = *(const v4h*)&feats_sh[k][c4];          // ds_load_b64
            a0 += s * (float)f[0]; a1 += s * (float)f[1];
            a2 += s * (float)f[2]; a3 += s * (float)f[3];
        }
        g_sh[wid * 128 + c4 + 0] = a0;
        g_sh[wid * 128 + c4 + 1] = a1;
        g_sh[wid * 128 + c4 + 2] = a2;
        g_sh[wid * 128 + c4 + 3] = a3;
    }
    __syncthreads();

    // ---- o = per-head g @ Wv, then x += o @ Wo ----
    if (t < 128) {
        const float* gp = &g_sh[(t >> 4) * 128];
        float a = 0.f;
        for (int c = 0; c < 128; ++c) a += gp[c] * Wv[c * 128 + t];
        o_sh[t] = a;
    }
    __syncthreads();
    if (t < 128) {
        float a = 0.f;
        for (int c = 0; c < 128; ++c) a += o_sh[c] * Wo[c * 128 + t];
        x_sh[t] += a;
    }
    __syncthreads();

    // ---- LN2 + FFN (gelu tanh approx, fast tanh) + residual ----
    block_ln_stats(x_sh, red_sh, t, mean, rstd);
    if (t < 128) h_sh[t] = (x_sh[t] - mean) * rstd * g2[t] + b2v[t];
    __syncthreads();

    for (int j = t; j < FFN_DIM; j += 256) {
        float a = b1f[j];
        for (int c = 0; c < 128; ++c) a += h_sh[c] * W1[c * FFN_DIM + j];
        const float u3 = a * a * a;
        const float inner = 0.7978845608f * (a + 0.044715f * u3);
        g_sh[j] = 0.5f * a * (1.f + fast_tanh(inner));
    }
    __syncthreads();

    // FFN down-projection: split-K across all 256 threads, combine via LDS
    {
        float* part = (float*)scores_sh;   // free after attention
        const int o = t & 127, half = t >> 7;
        float a = half ? 0.f : b2f[o];
        const int j0 = half * 256;
        for (int j = j0; j < j0 + 256; ++j) a += g_sh[j] * W2[j * 128 + o];
        part[t] = a;
        __syncthreads();
        if (t < 128) x_ws[n * 128 + t] = x_sh[t] + part[t] + part[t + 128];
    }
}

// Final logits: 7x7 level-0 patch, gathered with async-to-LDS (no format change needed).
__global__ __launch_bounds__(128)
void logits_kernel(const float* __restrict__ qpos, const int* __restrict__ boffs,
                   const float* __restrict__ fm, const int* __restrict__ lshapes,
                   const float* __restrict__ x_ws, float* __restrict__ out, int nb) {
    __shared__ float x_sh[128];
    __shared__ __align__(16) float patch_sh[49 * 128];
    __shared__ int roff_sh[49];
    __shared__ int ok_sh[49];

    const int n = blockIdx.x, t = threadIdx.x;
    if (t < 128) x_sh[t] = x_ws[n * 128 + t];
    const float px = qpos[2 * n + 0], py = qpos[2 * n + 1];
    int bid = 0;
    for (int b = 1; b < nb; ++b) if (n >= boffs[b]) bid = b;

    if (t < 49) {
        const int Hl = lshapes[0], Wl = lshapes[1];   // level 0 -> scale == 1
        const int di = t / 7 - 3, dj = t % 7 - 3;
        const int gi = (int)floorf(px) + di;
        const int gj = (int)floorf(py) + dj;
        const bool ok = (gi >= 0) && (gi < Hl) && (gj >= 0) && (gj < Wl);
        const int cci = gi < 0 ? 0 : (gi > Hl - 1 ? Hl - 1 : gi);
        const int ccj = gj < 0 ? 0 : (gj > Wl - 1 ? Wl - 1 : gj);
        roff_sh[t] = (((bid * 256 + cci) * 256 + ccj) * 4 + 0) * 128;
        ok_sh[t]   = ok ? 1 : 0;
    }
    __syncthreads();

#if USE_ASYNC_LOGITS
    // async DMA: 49 rows x 32 x b128 chunks straight into LDS (f32, no conversion)
    for (int c = t; c < 49 * 32; c += 128) {
        const int p = c >> 5, part = c & 31;
        const float* gp = fm + roff_sh[p] + part * 4;
        const unsigned loff = (unsigned)(uintptr_t)&patch_sh[p * 128 + part * 4];
        asm volatile("global_load_async_to_lds_b128 %0, %1, off"
                     :: "v"(loff), "v"(gp) : "memory");
    }
    asm volatile("s_wait_asynccnt 0x0" ::: "memory");
    __syncthreads();
#else
    for (int c = t; c < 49 * 32; c += 128) {
        const int p = c >> 5, part = c & 31;
        *(float4*)&patch_sh[p * 128 + part * 4] =
            *(const float4*)(fm + roff_sh[p] + part * 4);
    }
    __syncthreads();
#endif

    if (t < 49) {
        float r = 0.f;
        if (ok_sh[t]) {
            const float* fp = &patch_sh[t * 128];
            for (int c = 0; c < 128; ++c) r += x_sh[c] * fp[c];
        }
        out[n * 49 + t] = r;
    }
}

extern "C" void kernel_launch(void* const* d_in, const int* in_sizes, int n_in,
                              void* d_out, int out_size, void* d_ws, size_t ws_size,
                              hipStream_t stream) {
    const float* queries = (const float*)d_in[0];
    const int*   boffs   = (const int*)  d_in[1];
    const float* qpos    = (const float*)d_in[2];
    const float* fm      = (const float*)d_in[3];
    const int*   lsh     = (const int*)  d_in[4];
    const float* Wq      = (const float*)d_in[5];
    const float* Wk      = (const float*)d_in[6];
    const float* Wv      = (const float*)d_in[7];
    const float* Wo      = (const float*)d_in[8];
    const float* ln1g    = (const float*)d_in[9];
    const float* ln1b    = (const float*)d_in[10];
    const float* ln2g    = (const float*)d_in[11];
    const float* ln2b    = (const float*)d_in[12];
    const float* W1      = (const float*)d_in[13];
    const float* b1      = (const float*)d_in[14];
    const float* W2      = (const float*)d_in[15];
    const float* b2      = (const float*)d_in[16];

    const int N  = in_sizes[0] / D_MODEL;
    const int nb = in_sizes[1] - 1;

    float* x_ws = (float*)d_ws;   // residual stream: N x 128 f32
    hipMemcpyAsync(x_ws, queries, (size_t)N * D_MODEL * sizeof(float),
                   hipMemcpyDeviceToDevice, stream);

    for (int li = 0; li < 2; ++li) {
        layer_kernel<<<N, 256, 0, stream>>>(
            qpos, boffs, fm, lsh,
            Wq + (size_t)li * 128 * 128, Wk + (size_t)li * 128 * 128,
            Wv + (size_t)li * 128 * 128, Wo + (size_t)li * 128 * 128,
            ln1g + li * 128, ln1b + li * 128, ln2g + li * 128, ln2b + li * 128,
            W1 + (size_t)li * 128 * FFN_DIM, b1 + li * FFN_DIM,
            W2 + (size_t)li * FFN_DIM * 128, b2 + li * 128,
            x_ws, nb);
    }
    logits_kernel<<<N, 128, 0, stream>>>(qpos, boffs, fm, lsh, x_ws, (float*)d_out, nb);
}
